// RV2BEVFrustumAttn_15144054686409
// MI455X (gfx1250) — matrix operand order repaired and checked by reading, not compiled, not run on hardware
//
#include <hip/hip_runtime.h>
#include <hip/hip_bf16.h>
#include <math.h>

typedef __attribute__((ext_vector_type(16))) _Float16 v16h;
typedef __attribute__((ext_vector_type(8)))  float    v8f;

#define HRV   32
#define WRV   512
#define NPIX  (HRV*WRV)     // 16384
#define HB    256
#define WBV   256
#define NBEV  (HB*WBV)      // 65536
#define DMODEL 128
#define KTOP  5
#define NQ    (NPIX*KTOP)   // 81920
#define NHEADS 8
#define NPTS  6
#define HD    16
#define NBINS 110

__device__ __constant__ float c_elev_deg[32] = {
  -30.67f,-29.33f,-28.0f,-26.66f,-25.33f,-24.0f,-22.67f,-21.33f,
  -20.0f,-18.67f,-17.33f,-16.0f,-14.67f,-13.33f,-12.0f,-10.67f,
  -9.33f,-8.0f,-6.66f,-5.33f,-4.0f,-2.67f,-1.33f,0.0f,1.33f,
  2.67f,4.0f,5.33f,6.67f,8.0f,9.33f,10.67f};

__device__ __forceinline__ float gelu_exact(float v) {
  return 0.5f * v * (1.0f + erff(v * 0.70710678118654752f));
}

// ---------------- unit ray vectors (host constants recomputed on device) ----
__global__ void k_uvec(float* __restrict__ ux, float* __restrict__ uy, float* __restrict__ uz) {
  int m = blockIdx.x * blockDim.x + threadIdx.x;
  if (m >= NPIX) return;
  int h = m >> 9, w = m & 511;
  const float PI = 3.14159265358979323846f;
  float az = -PI + ((float)w + 0.5f) * (2.0f * PI / 512.0f);
  float el = c_elev_deg[31 - h] * (PI / 180.0f);
  float ce = cosf(el);
  ux[m] = cosf(az) * ce;
  uy[m] = sinf(az) * ce;
  uz[m] = sinf(el);
}

__global__ void k_zero(float* p, int n) {
  int i = blockIdx.x * blockDim.x + threadIdx.x;
  if (i < n) p[i] = 0.0f;
}

// ---------------- generic WMMA GEMM: C[M,N] = act(A[M,K] * W[N,K]^T + b) ----
// AMODE: 0 = A row-major [M,K]
//        1 = A channel-major [K,M]   (NCHW bev -> NHWC linear)
//        2 = A = concat(x_rv[M,64], ux, uy, uz)  (K=67)
//        3 = A = im2col of h1[NPIX,128] with 3x3 circular taps (K=1152)
// WMODE: 0 = W row-major [N,K];  1 = conv OIHW [N,128,3,3], k = tap*128+c
// ACT:   0 = none, 1 = exact GELU
// VEC4:  float4 staging path (requires AMODE==0, WMODE==0, K%32==0)
template <int AMODE, int WMODE, int ACT, bool RESID, bool VEC4>
__global__ void k_gemm(const float* __restrict__ A, const float* __restrict__ W,
                       const float* __restrict__ bias, float* __restrict__ C,
                       const float* __restrict__ resid,
                       const float* __restrict__ aux1, const float* __restrict__ aux2,
                       const float* __restrict__ aux3,
                       int M, int N, int K) {
  __shared__ float lA[16][36];   // pad 36: keeps float4 LDS stores 16B-aligned
  __shared__ float lB[16][36];
  int lane  = threadIdx.x;
  int mtile = blockIdx.x << 4;
  int ntile = blockIdx.y << 4;
  v8f acc = {};

  for (int kc = 0; kc < K; kc += 32) {
    if constexpr (VEC4) {
      // 16x32 chunk = 128 float4 units; 4 per lane. K%32==0 -> no tail guard.
#pragma unroll
      for (int t = 0; t < 4; ++t) {
        int i  = t * 32 + lane;
        int r  = i >> 3;
        int kk = (i & 7) << 2;
        int k  = kc + kk;
        float4 av = *(const float4*)&A[(size_t)(mtile + r) * K + k];
        *(float4*)&lA[r][kk] = av;
        if (kc + 32 < K)
          __builtin_prefetch(&A[(size_t)(mtile + r) * K + k + 32], 0, 1);
        int n = ntile + r;
        float4 wv = make_float4(0.f, 0.f, 0.f, 0.f);
        if (n < N) wv = *(const float4*)&W[(size_t)n * K + k];
        *(float4*)&lB[r][kk] = wv;
      }
    } else {
      for (int i = lane; i < 512; i += 32) {
        int r, kk;
        if constexpr (AMODE == 1) { r = i & 15; kk = i >> 4; }  // lanes sweep m
        else                      { r = i >> 5; kk = i & 31; }  // lanes sweep k
        int k = kc + kk;
        float av = 0.0f, wv = 0.0f;
        if (k < K) {
          int m = mtile + r;
          if constexpr (AMODE == 0) {
            av = A[(size_t)m * K + k];
          } else if constexpr (AMODE == 1) {
            av = A[(size_t)k * M + m];
          } else if constexpr (AMODE == 2) {
            if      (k < 64)  av = A[(size_t)m * 64 + k];
            else if (k == 64) av = aux1[m];
            else if (k == 65) av = aux2[m];
            else              av = aux3[m];
          } else {  // 3x3 circular conv im2col on the fly
            int tap = k >> 7, c = k & 127;
            int dh = tap / 3 - 1, dw = tap - (tap / 3) * 3 - 1;
            int h = ((m >> 9) + dh + HRV) & (HRV - 1);
            int w = ((m & 511) + dw + WRV) & (WRV - 1);
            av = A[(size_t)(h * WRV + w) * 128 + c];
          }
          int n = ntile + r;
          if (n < N) {
            if constexpr (WMODE == 0) {
              wv = W[(size_t)n * K + k];
            } else {
              int tap = k >> 7, c = k & 127;
              wv = W[(size_t)n * 1152 + c * 9 + tap];
            }
          }
        }
        lA[r][kk] = av;
        lB[r][kk] = wv;
      }
    }
    __syncthreads();

    // build fragments per CDNA5 wave32 WMMA layouts
    int mr = lane & 15, sel = lane >> 4;
    v16h a, b;
#pragma unroll
    for (int j = 0; j < 8; ++j) {
      int ka = ((j & 4) ? 16 : 0) + sel * 8 + ((j & 3) << 1);
      a[2 * j]     = (_Float16)lA[mr][ka];
      a[2 * j + 1] = (_Float16)lA[mr][ka + 1];
      int kb = ((j & 4) ? 16 : 0) + sel * 8 + ((j & 3) << 1);
      b[2 * j]     = (_Float16)lB[mr][kb];
      b[2 * j + 1] = (_Float16)lB[mr][kb + 1];
    }
    acc = __builtin_amdgcn_wmma_f32_16x16x32_f16(false, a, false, b, (short)0, acc, false, false);
    __syncthreads();
  }

  int nc = ntile + (lane & 15);
  if (nc < N) {
    float bv = bias ? bias[nc] : 0.0f;
    int sel = lane >> 4;
#pragma unroll
    for (int r = 0; r < 8; ++r) {
      int mo = mtile + sel * 8 + r;
      float v = acc[r] + bv;
      if constexpr (RESID) v += resid[(size_t)mo * N + nc];
      if constexpr (ACT == 1) v = gelu_exact(v);
      C[(size_t)mo * N + nc] = v;
    }
  }
}

// ---------------- GroupNorm (groups=8) global reduce + apply + GELU ---------
__global__ void k_gn_reduce(const float* __restrict__ x, float* __restrict__ acc,
                            int M, int C) {
  __shared__ float ssum[8], ssq[8];
  int cpg = C >> 3;
  if (threadIdx.x < 8) { ssum[threadIdx.x] = 0.0f; ssq[threadIdx.x] = 0.0f; }
  __syncthreads();
  size_t total = (size_t)M * C;
  for (size_t i = (size_t)blockIdx.x * blockDim.x + threadIdx.x; i < total;
       i += (size_t)gridDim.x * blockDim.x) {
    int c = (int)(i % C);
    int g = c / cpg;
    float v = x[i];
    atomicAdd(&ssum[g], v);
    atomicAdd(&ssq[g], v * v);
  }
  __syncthreads();
  if (threadIdx.x < 8) {
    atomicAdd(&acc[threadIdx.x], ssum[threadIdx.x]);
    atomicAdd(&acc[8 + threadIdx.x], ssq[threadIdx.x]);
  }
}

__global__ void k_gn_apply(float* __restrict__ x, const float* __restrict__ acc,
                           const float* __restrict__ gamma, const float* __restrict__ beta,
                           int M, int C) {
  size_t i = (size_t)blockIdx.x * blockDim.x + threadIdx.x;
  if (i >= (size_t)M * C) return;
  int c = (int)(i % C);
  int cpg = C >> 3;
  int g = c / cpg;
  float cnt = (float)M * (float)cpg;
  float mu  = acc[g] / cnt;
  float var = acc[8 + g] / cnt - mu * mu;
  float v = (x[i] - mu) * rsqrtf(var + 1e-5f) * gamma[c] + beta[c];
  x[i] = gelu_exact(v);
}

// ---------------- softmax over 110 bins + top-5 (depth + norm weight) -------
__global__ void k_topk(const float* __restrict__ dl, float* __restrict__ tdep,
                       float* __restrict__ wgt) {
  int m = blockIdx.x * blockDim.x + threadIdx.x;
  if (m >= NPIX) return;
  const float* row = dl + (size_t)m * NBINS;
  float mx = row[0];
  for (int c = 1; c < NBINS; ++c) mx = fmaxf(mx, row[c]);
  float s = 0.0f;
  for (int c = 0; c < NBINS; ++c) s += expf(row[c] - mx);
  int bi[KTOP];
  float pv[KTOP];
  for (int k = 0; k < KTOP; ++k) {
    float best = -1e30f; int bidx = 0;
    for (int c = 0; c < NBINS; ++c) {
      bool skip = false;
      for (int j = 0; j < k; ++j) if (bi[j] == c) skip = true;
      float l = row[c];
      if (!skip && l > best) { best = l; bidx = c; }
    }
    bi[k] = bidx;
    pv[k] = expf(best - mx) / s;
  }
  float ws = 1e-8f;
  for (int k = 0; k < KTOP; ++k) ws += pv[k];
  for (int k = 0; k < KTOP; ++k) {
    wgt[(size_t)m * KTOP + k]  = pv[k] / ws;
    tdep[(size_t)m * KTOP + k] = fminf(0.5f * ((float)bi[k] + 0.5f), 54.75f);
  }
}

// ---------------- frustum point projection -> normalized BEV ref points -----
__global__ void k_refpts(const float* __restrict__ tdep, const float* __restrict__ ux,
                         const float* __restrict__ uy, const float* __restrict__ uz,
                         const float* __restrict__ L, float* __restrict__ refp) {
  int q = blockIdx.x * blockDim.x + threadIdx.x;
  if (q >= NQ) return;
  int m = q / KTOP;
  float d = tdep[q];
  float px = d * ux[m], py = d * uy[m], pz = d * uz[m];
  float pe0 = px * L[0] + py * L[4] + pz * L[8]  + L[12];
  float pe1 = px * L[1] + py * L[5] + pz * L[9]  + L[13];
  float rx = (pe0 + 55.0f) * (1.0f / 110.0f);
  float ry = (pe1 + 55.0f) * (1.0f / 110.0f);
  refp[(size_t)q * 2]     = fminf(fmaxf(rx, 0.0f), 1.0f);
  refp[(size_t)q * 2 + 1] = fminf(fmaxf(ry, 0.0f), 1.0f);
}

// ---------------- assemble qin = [Q0 broadcast over K, depth/RMAX] ----------
__global__ void k_qin(const float* __restrict__ Q0, const float* __restrict__ tdep,
                      float* __restrict__ qin) {
  size_t i = (size_t)blockIdx.x * blockDim.x + threadIdx.x;
  if (i >= (size_t)NQ * 129) return;
  int q = (int)(i / 129);
  int c = (int)(i % 129);
  int m = q / KTOP;
  qin[i] = (c < DMODEL) ? Q0[(size_t)m * DMODEL + c] : tdep[q] * (1.0f / 55.0f);
}

// ---------------- deformable bilinear sampling (one thread per q,head) ------
__global__ void k_msda(const float* __restrict__ vproj, const float* __restrict__ offb,
                       const float* __restrict__ awb, const float* __restrict__ refp,
                       float* __restrict__ samp) {
  int t = blockIdx.x * blockDim.x + threadIdx.x;
  if (t >= NQ * NHEADS) return;
  int q = t >> 3, head = t & 7;
  const float* offq = offb + (size_t)q * (NHEADS * NPTS * 2) + head * (NPTS * 2);
  const float* awq  = awb  + (size_t)q * (NHEADS * NPTS)     + head * NPTS;
  float amx = awq[0];
  for (int p = 1; p < NPTS; ++p) amx = fmaxf(amx, awq[p]);
  float ae[NPTS], asum = 0.0f;
  for (int p = 0; p < NPTS; ++p) { ae[p] = expf(awq[p] - amx); asum += ae[p]; }
  float rx = refp[(size_t)q * 2], ry = refp[(size_t)q * 2 + 1];
  float accv[HD];
#pragma unroll
  for (int d = 0; d < HD; ++d) accv[d] = 0.0f;
  for (int p = 0; p < NPTS; ++p) {
    float wa = ae[p] / asum;
    float x = (rx + offq[p * 2]     * (1.0f / (float)WBV)) * (float)WBV - 0.5f;
    float y = (ry + offq[p * 2 + 1] * (1.0f / (float)HB))  * (float)HB  - 0.5f;
    float x0 = floorf(x), y0 = floorf(y);
    float wx = x - x0, wy = y - y0;
    int xi = (int)x0, yi = (int)y0;
#pragma unroll
    for (int cy = 0; cy < 2; ++cy) {
#pragma unroll
      for (int cx = 0; cx < 2; ++cx) {
        int xx = xi + cx, yy = yi + cy;
        float wc = (cx ? wx : 1.0f - wx) * (cy ? wy : 1.0f - wy);
        if (xx >= 0 && xx < WBV && yy >= 0 && yy < HB) {
          const float* vp = vproj + ((size_t)(yy * WBV + xx) * DMODEL + head * HD);
          float wf = wa * wc;
#pragma unroll
          for (int d = 0; d < HD; ++d) accv[d] += vp[d] * wf;
        }
      }
    }
  }
  float* o = samp + (size_t)q * DMODEL + head * HD;
#pragma unroll
  for (int d = 0; d < HD; ++d) o[d] = accv[d];
}

// ---------------- weighted sum over top-K candidates ------------------------
__global__ void k_wsum(const float* __restrict__ ymsda, const float* __restrict__ wgt,
                       float* __restrict__ ybev) {
  size_t i = (size_t)blockIdx.x * blockDim.x + threadIdx.x;
  if (i >= (size_t)NPIX * DMODEL) return;
  int m = (int)(i >> 7);
  int c = (int)(i & 127);
  float s = 0.0f;
#pragma unroll
  for (int k = 0; k < KTOP; ++k)
    s += ymsda[((size_t)m * KTOP + k) * DMODEL + c] * wgt[(size_t)m * KTOP + k];
  ybev[i] = s;
}

// ---------------- depth logits transpose [NPIX,NBINS] -> [NBINS,NPIX] -------
__global__ void k_tlog(const float* __restrict__ dl, float* __restrict__ out) {
  size_t i = (size_t)blockIdx.x * blockDim.x + threadIdx.x;
  if (i >= (size_t)NPIX * NBINS) return;
  int m = (int)(i / NBINS);
  int c = (int)(i % NBINS);
  out[(size_t)c * NPIX + m] = dl[i];
}

// ============================================================================
extern "C" void kernel_launch(void* const* d_in, const int* in_sizes, int n_in,
                              void* d_out, int out_size, void* d_ws, size_t ws_size,
                              hipStream_t stream) {
  (void)in_sizes; (void)n_in; (void)out_size; (void)ws_size;
  const float* x_rv  = (const float*)d_in[0];
  const float* bev   = (const float*)d_in[1];
  const float* Lm    = (const float*)d_in[2];
  const float* wq    = (const float*)d_in[3];
  const float* bq    = (const float*)d_in[4];
  const float* wv    = (const float*)d_in[5];
  const float* bv    = (const float*)d_in[6];
  const float* wo    = (const float*)d_in[7];
  const float* bo    = (const float*)d_in[8];
  const float* qd_w1 = (const float*)d_in[9];
  const float* qd_b1 = (const float*)d_in[10];
  const float* qd_w2 = (const float*)d_in[11];
  const float* qd_b2 = (const float*)d_in[12];
  const float* rh_w1 = (const float*)d_in[13];
  const float* rh_b1 = (const float*)d_in[14];
  const float* rh_g1 = (const float*)d_in[15];
  const float* rh_be1= (const float*)d_in[16];
  const float* rh_w2 = (const float*)d_in[17];
  const float* rh_g2 = (const float*)d_in[18];
  const float* rh_be2= (const float*)d_in[19];
  const float* rh_w3 = (const float*)d_in[20];
  const float* rh_b3 = (const float*)d_in[21];
  const float* off_w = (const float*)d_in[22];
  const float* off_b = (const float*)d_in[23];
  const float* aw_w  = (const float*)d_in[24];
  const float* aw_b  = (const float*)d_in[25];
  const float* vp_w  = (const float*)d_in[26];
  const float* vp_b  = (const float*)d_in[27];
  const float* op_w  = (const float*)d_in[28];
  const float* op_b  = (const float*)d_in[29];

  float* ws = (float*)d_ws;
  size_t off = 0;
  auto alloc = [&](size_t n) { float* p = ws + off; off += n; return p; };
  float* ux    = alloc(NPIX);
  float* uy    = alloc(NPIX);
  float* uz    = alloc(NPIX);
  float* gnacc = alloc(32);              // [0..15] GN1, [16..31] GN2
  float* Q0    = alloc((size_t)NPIX * DMODEL);
  float* Vmap  = alloc((size_t)NBEV * DMODEL);
  float* h1    = alloc((size_t)NPIX * 128);
  float* h2    = alloc((size_t)NPIX * 64);
  float* dl    = alloc((size_t)NPIX * NBINS);
  float* tdep  = alloc((size_t)NPIX * KTOP);
  float* wgt   = alloc((size_t)NPIX * KTOP);
  float* refp  = alloc((size_t)NQ * 2);
  float* qin   = alloc((size_t)NQ * 129);
  float* hq    = alloc((size_t)NQ * DMODEL);
  float* query = alloc((size_t)NQ * DMODEL);
  float* vproj = alloc((size_t)NBEV * DMODEL);
  float* offb  = alloc((size_t)NQ * NHEADS * NPTS * 2);
  float* awb   = alloc((size_t)NQ * NHEADS * NPTS);
  float* ybev  = alloc((size_t)NPIX * DMODEL);
  float* samp  = qin;   // alias: qin dead after qd1
  float* ymsda = hq;    // alias: hq dead after qd2

  float* ydst = (float*)d_out;
  float* dlt  = ydst + (size_t)NPIX * DMODEL;

#define GEMM(AM, WM, ACT, RES, VEC, Ap, Wp, Bp, Cp, Rp, a1, a2, a3, Mv, Nv, Kv)      \
  k_gemm<AM, WM, ACT, RES, VEC>                                                       \
      <<<dim3((Mv) / 16, ((Nv) + 15) / 16), 32, 0, stream>>>(                         \
          P_(Ap), Wp, Bp, Cp, Rp, a1, a2, a3, Mv, Nv, Kv)
#define P_(x) x

  k_uvec<<<(NPIX + 255) / 256, 256, 0, stream>>>(ux, uy, uz);
  k_zero<<<1, 32, 0, stream>>>(gnacc, 32);

  // Q0 = x_rv @ wq^T + bq                       (vec4, K=64)
  GEMM(0, 0, 0, false, true, x_rv, wq, bq, Q0, nullptr, nullptr, nullptr, nullptr,
       NPIX, DMODEL, 64);
  // Vmap = NHWC(bev) @ wv^T + bv                (channel-major A)
  GEMM(1, 0, 0, false, false, bev, wv, bv, Vmap, nullptr, nullptr, nullptr, nullptr,
       NBEV, DMODEL, 128);
  // h1 = [x_rv ; uvec] @ rh_w1^T + rh_b1        (concat A, K=67)
  GEMM(2, 0, 0, false, false, x_rv, rh_w1, rh_b1, h1, nullptr, ux, uy, uz,
       NPIX, 128, 67);

  k_gn_reduce<<<256, 256, 0, stream>>>(h1, gnacc, NPIX, 128);
  k_gn_apply<<<((size_t)NPIX * 128 + 255) / 256, 256, 0, stream>>>(h1, gnacc, rh_g1, rh_be1, NPIX, 128);

  // 3x3 circular conv as WMMA GEMM, K = 9*128 with im2col on the fly
  GEMM(3, 1, 0, false, false, h1, rh_w2, (const float*)nullptr, h2, nullptr,
       nullptr, nullptr, nullptr, NPIX, 64, 1152);
  k_gn_reduce<<<256, 256, 0, stream>>>(h2, gnacc + 16, NPIX, 64);
  k_gn_apply<<<((size_t)NPIX * 64 + 255) / 256, 256, 0, stream>>>(h2, gnacc + 16, rh_g2, rh_be2, NPIX, 64);

  // depth logits                                (vec4, K=64, N=110 guarded)
  GEMM(0, 0, 0, false, true, h2, rh_w3, rh_b3, dl, nullptr, nullptr, nullptr, nullptr,
       NPIX, NBINS, 64);
  k_topk<<<(NPIX + 255) / 256, 256, 0, stream>>>(dl, tdep, wgt);
  k_refpts<<<(NQ + 255) / 256, 256, 0, stream>>>(tdep, ux, uy, uz, Lm, refp);
  k_qin<<<((size_t)NQ * 129 + 255) / 256, 256, 0, stream>>>(Q0, tdep, qin);

  // query MLP: GELU(qin @ qd_w1^T + b) @ qd_w2^T + b
  GEMM(0, 0, 1, false, false, qin, qd_w1, qd_b1, hq, nullptr, nullptr, nullptr, nullptr,
       NQ, DMODEL, 129);
  GEMM(0, 0, 0, false, true, hq, qd_w2, qd_b2, query, nullptr, nullptr, nullptr, nullptr,
       NQ, DMODEL, 128);

  // value projection, offsets, attention-weight logits (all vec4, K=128)
  GEMM(0, 0, 0, false, true, Vmap, vp_w, vp_b, vproj, nullptr, nullptr, nullptr, nullptr,
       NBEV, DMODEL, 128);
  GEMM(0, 0, 0, false, true, query, off_w, off_b, offb, nullptr, nullptr, nullptr, nullptr,
       NQ, NHEADS * NPTS * 2, 128);
  GEMM(0, 0, 0, false, true, query, aw_w, aw_b, awb, nullptr, nullptr, nullptr, nullptr,
       NQ, NHEADS * NPTS, 128);

  // deformable bilinear sampling (+ per-head softmax over P inline)
  k_msda<<<(NQ * NHEADS + 255) / 256, 256, 0, stream>>>(vproj, offb, awb, refp, samp);

  // output projection + residual (vec4 + residual epilogue)
  GEMM(0, 0, 0, true, true, samp, op_w, op_b, ymsda, query, nullptr, nullptr, nullptr,
       NQ, DMODEL, 128);

  // weighted reduction over top-K, then final projection straight into d_out
  k_wsum<<<((size_t)NPIX * DMODEL + 255) / 256, 256, 0, stream>>>(ymsda, wgt, ybev);
  GEMM(0, 0, 0, false, true, ybev, wo, bo, ydst, nullptr, nullptr, nullptr, nullptr,
       NPIX, DMODEL, 128);

  // depth logits transposed to second output region
  k_tlog<<<((size_t)NPIX * NBINS + 255) / 256, 256, 0, stream>>>(dl, dlt);

#undef GEMM
#undef P_
}
